// ContraNorm_12730283065572
// MI455X (gfx1250) — compile-verified, matrix operand order checked by reading, not run on hardware
//
#include <hip/hip_runtime.h>
#include <hip/hip_bf16.h>

typedef __bf16 bf16_t;
typedef __attribute__((ext_vector_type(16))) __bf16 v16bf;
typedef __attribute__((ext_vector_type(2)))  __bf16 v2bf;
typedef __attribute__((ext_vector_type(8)))  float  v8f;

#define CN_SCALE    0.1f
#define CN_LN_EPS   1e-6f
#define CN_NORM_EPS 1e-12f

// ---------------------------------------------------------------------------
// WMMA fragment helper (v_wmma_f32_16x16x32_bf16, wave32)
// ISA 7.12.2, 16-bit A 16x32: lanes 0-15 carry rows M=0-15 (lanes 16-31 same
// rows, second K-half); bf16 pair j of the fragment holds
//   K = (j<4 ? 8h + 2j : 16 + 8h + 2(j-4)) , +1      (h = lane>=16)
// Per lane this is two contiguous 16B chunks -> compiler emits 2x b128 loads.
// ---------------------------------------------------------------------------
static __device__ __forceinline__ int kpair_base(int j, int h) {
  return (j < 4) ? (8 * h + 2 * j) : (16 + 8 * h + 2 * (j - 4));
}

// Row-major source, lane supplies row (row0 + lane%16), K contiguous in memory.
// Used for: A operands; B of xn@xn^T (symmetric gather); B of dual@x via xT.
static __device__ __forceinline__ v16bf frag_rowmajor(const bf16_t* __restrict__ p,
                                                      int ld, int row0, int k0, int lane) {
  const bf16_t* r = p + (size_t)(row0 + (lane & 15)) * ld + k0;
  const int h = lane >> 4;
  v16bf f;
#pragma unroll
  for (int j = 0; j < 8; ++j) {
    v2bf d = *(const v2bf*)(r + kpair_base(j, h));   // 4B-aligned
    f[2 * j]     = d[0];
    f[2 * j + 1] = d[1];
  }
  return f;
}

#define WMMA_BF16(a, b, c) \
  __builtin_amdgcn_wmma_f32_16x16x32_bf16(false, (a), false, (b), (short)0, (c), false, false)

// ---------------------------------------------------------------------------
// 1) L2-normalize rows -> xn (bf16). One 256-thread block per row (D = 256).
// ---------------------------------------------------------------------------
__global__ __launch_bounds__(256) void k_normalize_cast(const float* __restrict__ x,
                                                        bf16_t* __restrict__ xn, int D) {
  __shared__ float sm[256];
  const int row = blockIdx.x, t = threadIdx.x;
  const size_t i = (size_t)row * D + t;
  const float v = x[i];
  sm[t] = v * v;
  __syncthreads();
  for (int o = 128; o > 0; o >>= 1) {
    if (t < o) sm[t] += sm[t + o];
    __syncthreads();
  }
  const float inv = 1.0f / fmaxf(__builtin_sqrtf(sm[0]), CN_NORM_EPS);
  xn[i] = (bf16_t)(v * inv);
}

// ---------------------------------------------------------------------------
// 1b) xT[b][d][n] = bf16(x[b][n][d]) — LDS-tiled transpose, coalesced both ways.
//     Makes GEMM2's B operand a contiguous-K (vectorizable) gather.
// ---------------------------------------------------------------------------
__global__ __launch_bounds__(256) void k_transpose_cast(const float* __restrict__ x,
                                                        bf16_t* __restrict__ xT,
                                                        int N, int D) {
  __shared__ bf16_t tile[32][33];                    // +1 pad: no bank conflicts
  const float* xb = x + (size_t)blockIdx.z * N * D;
  bf16_t* xTb = xT + (size_t)blockIdx.z * N * D;
  const int r0 = blockIdx.y * 32;                    // n
  const int c0 = blockIdx.x * 32;                    // d
  const int tx = threadIdx.x & 31, ty = threadIdx.x >> 5;
#pragma unroll
  for (int i = 0; i < 32; i += 8)
    tile[ty + i][tx] = (bf16_t)xb[(size_t)(r0 + ty + i) * D + c0 + tx];
  __syncthreads();
#pragma unroll
  for (int i = 0; i < 32; i += 8)
    xTb[(size_t)(c0 + ty + i) * N + r0 + tx] = tile[tx][ty + i];
}

// ---------------------------------------------------------------------------
// 2) sim = xn @ xn^T  (N x N, K = D).
//    Wave: 32x64 (2 A-frags x 4 B-frags, 8 f32 accums). Block 8 waves: 128x128.
// ---------------------------------------------------------------------------
__global__ __launch_bounds__(256) void k_gemm_sim(const bf16_t* __restrict__ xn,
                                                  float* __restrict__ sim, int N, int D) {
  const int lane = threadIdx.x & 31;
  const int w  = threadIdx.x >> 5;
  const int wr = w >> 1, wc = w & 1;                 // 4x2 wave grid
  const int m0 = blockIdx.y * 128 + wr * 32;
  const int n0 = blockIdx.x * 128 + wc * 64;
  v8f c00 = {}, c01 = {}, c02 = {}, c03 = {};
  v8f c10 = {}, c11 = {}, c12 = {}, c13 = {};
  for (int k0 = 0; k0 < D; k0 += 32) {
    if (k0 + 32 < D)                                 // global_prefetch_b8
      __builtin_prefetch(xn + (size_t)(m0 + (lane & 15)) * D + k0 + 32, 0, 1);
    v16bf a0 = frag_rowmajor(xn, D, m0,      k0, lane);
    v16bf a1 = frag_rowmajor(xn, D, m0 + 16, k0, lane);
    v16bf b0 = frag_rowmajor(xn, D, n0,      k0, lane);  // B = xn^T: same gather
    v16bf b1 = frag_rowmajor(xn, D, n0 + 16, k0, lane);
    v16bf b2 = frag_rowmajor(xn, D, n0 + 32, k0, lane);
    v16bf b3 = frag_rowmajor(xn, D, n0 + 48, k0, lane);
    c00 = WMMA_BF16(a0, b0, c00);  c01 = WMMA_BF16(a0, b1, c01);
    c02 = WMMA_BF16(a0, b2, c02);  c03 = WMMA_BF16(a0, b3, c03);
    c10 = WMMA_BF16(a1, b0, c10);  c11 = WMMA_BF16(a1, b1, c11);
    c12 = WMMA_BF16(a1, b2, c12);  c13 = WMMA_BF16(a1, b3, c13);
  }
  const int h = lane >> 4, ln = lane & 15;
#pragma unroll
  for (int g = 0; g < 8; ++g) {
    const int mA = m0 + g + 8 * h;                   // TEMP == 1.0
    const int mB = mA + 16;
    float* rA = sim + (size_t)mA * N + n0 + ln;
    float* rB = sim + (size_t)mB * N + n0 + ln;
    rA[0] = c00[g]; rA[16] = c01[g]; rA[32] = c02[g]; rA[48] = c03[g];
    rB[0] = c10[g]; rB[16] = c11[g]; rB[32] = c12[g]; rB[48] = c13[g];
  }
}

// ---------------------------------------------------------------------------
// 3) Row softmax stats (sim symmetric -> column stats == row stats).
// ---------------------------------------------------------------------------
__global__ __launch_bounds__(256) void k_row_stats(const float* __restrict__ sim,
                                                   float* __restrict__ rmax,
                                                   float* __restrict__ rsuminv, int N) {
  __shared__ float sm[256];
  const int row = blockIdx.x, t = threadIdx.x;
  const float* r = sim + (size_t)row * N;
  float m = -3.4e38f;
  for (int i = t; i < N; i += 256) m = fmaxf(m, r[i]);
  sm[t] = m;
  __syncthreads();
  for (int o = 128; o > 0; o >>= 1) {
    if (t < o) sm[t] = fmaxf(sm[t], sm[t + o]);
    __syncthreads();
  }
  const float M = sm[0];
  __syncthreads();
  float s = 0.f;
  for (int i = t; i < N; i += 256) s += __expf(r[i] - M);
  sm[t] = s;
  __syncthreads();
  for (int o = 128; o > 0; o >>= 1) {
    if (t < o) sm[t] += sm[t + o];
    __syncthreads();
  }
  if (t == 0) {
    rmax[row] = M;
    rsuminv[row] = 1.0f / sm[0];
  }
}

// ---------------------------------------------------------------------------
// 4) dual[n,m] = exp(s - rmax[n])/rsum[n] + exp(s - rmax[m])/rsum[m]  -> bf16
// ---------------------------------------------------------------------------
__global__ __launch_bounds__(256) void k_dualize(const float* __restrict__ sim,
                                                 const float* __restrict__ rmax,
                                                 const float* __restrict__ rsuminv,
                                                 bf16_t* __restrict__ dual, int N) {
  const size_t idx = (size_t)blockIdx.x * 256 + threadIdx.x;
  const int n = (int)(idx / N), m = (int)(idx % N);    // N = 2^11 -> shifts
  const float s = sim[idx];
  const float v = __expf(s - rmax[n]) * rsuminv[n] + __expf(s - rmax[m]) * rsuminv[m];
  dual[idx] = (bf16_t)v;
}

// ---------------------------------------------------------------------------
// 5) x_neg = dual @ x  (N x D, K = N).  B fragment comes from xT so the gather
//    is contiguous-K (b128 loads). Wave 32x64; block 8 waves: 128x128.
// ---------------------------------------------------------------------------
__global__ __launch_bounds__(256) void k_gemm_neg(const bf16_t* __restrict__ dual,
                                                  const bf16_t* __restrict__ xT,
                                                  float* __restrict__ xneg, int N, int D) {
  const int lane = threadIdx.x & 31;
  const int w  = threadIdx.x >> 5;
  const int wr = w >> 1, wc = w & 1;
  const int m0 = blockIdx.y * 128 + wr * 32;
  const int j0 = blockIdx.x * 128 + wc * 64;
  v8f c00 = {}, c01 = {}, c02 = {}, c03 = {};
  v8f c10 = {}, c11 = {}, c12 = {}, c13 = {};
  for (int k0 = 0; k0 < N; k0 += 32) {
    if (k0 + 32 < N)
      __builtin_prefetch(dual + (size_t)(m0 + (lane & 15)) * N + k0 + 32, 0, 1);
    v16bf a0 = frag_rowmajor(dual, N, m0,      k0, lane);
    v16bf a1 = frag_rowmajor(dual, N, m0 + 16, k0, lane);
    v16bf b0 = frag_rowmajor(xT, N, j0,      k0, lane);  // B[k][j] = xT[j][k]
    v16bf b1 = frag_rowmajor(xT, N, j0 + 16, k0, lane);
    v16bf b2 = frag_rowmajor(xT, N, j0 + 32, k0, lane);
    v16bf b3 = frag_rowmajor(xT, N, j0 + 48, k0, lane);
    c00 = WMMA_BF16(a0, b0, c00);  c01 = WMMA_BF16(a0, b1, c01);
    c02 = WMMA_BF16(a0, b2, c02);  c03 = WMMA_BF16(a0, b3, c03);
    c10 = WMMA_BF16(a1, b0, c10);  c11 = WMMA_BF16(a1, b1, c11);
    c12 = WMMA_BF16(a1, b2, c12);  c13 = WMMA_BF16(a1, b3, c13);
  }
  const int h = lane >> 4, ln = lane & 15;
#pragma unroll
  for (int g = 0; g < 8; ++g) {
    const int mA = m0 + g + 8 * h;
    const int mB = mA + 16;
    float* rA = xneg + (size_t)mA * D + j0 + ln;
    float* rB = xneg + (size_t)mB * D + j0 + ln;
    rA[0] = c00[g]; rA[16] = c01[g]; rA[32] = c02[g]; rA[48] = c03[g];
    rB[0] = c10[g]; rB[16] = c11[g]; rB[32] = c12[g]; rB[48] = c13[g];
  }
}

// ---------------------------------------------------------------------------
// 6) out = LayerNorm(x - SCALE * x_neg) * gamma + beta.  One block per row.
// ---------------------------------------------------------------------------
__global__ __launch_bounds__(256) void k_residual_ln(const float* __restrict__ x,
                                                     const float* __restrict__ xneg,
                                                     const float* __restrict__ gamma,
                                                     const float* __restrict__ beta,
                                                     float* __restrict__ out, int D) {
  __shared__ float s1[256];
  __shared__ float s2[256];
  const int row = blockIdx.x, t = threadIdx.x;
  const size_t i = (size_t)row * D + t;
  const float v = x[i] - CN_SCALE * xneg[i];
  s1[t] = v;
  s2[t] = v * v;
  __syncthreads();
  for (int o = 128; o > 0; o >>= 1) {
    if (t < o) { s1[t] += s1[t + o]; s2[t] += s2[t + o]; }
    __syncthreads();
  }
  const float mu = s1[0] / D;
  const float var = s2[0] / D - mu * mu;
  const float rs = __builtin_amdgcn_rsqf(var + CN_LN_EPS);
  out[i] = (v - mu) * rs * gamma[t] + beta[t];
}

// ---------------------------------------------------------------------------
extern "C" void kernel_launch(void* const* d_in, const int* in_sizes, int n_in,
                              void* d_out, int out_size, void* d_ws, size_t ws_size,
                              hipStream_t stream) {
  const int B = 8, N = 2048, D = 256;
  const float* x     = (const float*)d_in[0];
  const float* gamma = (const float*)d_in[1];
  const float* beta  = (const float*)d_in[2];
  float* out = (float*)d_out;

  char* ws = (char*)d_ws;
  bf16_t* xn   = (bf16_t*)ws;  ws += (size_t)B * N * D * sizeof(bf16_t);  // 8.4 MB
  bf16_t* xT   = (bf16_t*)ws;  ws += (size_t)B * N * D * sizeof(bf16_t);  // 8.4 MB ([B][D][N])
  float*  sim  = (float*)ws;   ws += (size_t)N * N * sizeof(float);       // 16.8 MB (per-batch reuse)
  bf16_t* dual = (bf16_t*)ws;  ws += (size_t)N * N * sizeof(bf16_t);      // 8.4 MB (per-batch reuse)
  float*  rmax = (float*)ws;   ws += (size_t)N * sizeof(float);
  float*  rsmi = (float*)ws;   ws += (size_t)N * sizeof(float);
  float*  xneg = (float*)ws;                                              // 2.1 MB (per-batch reuse)

  k_normalize_cast<<<B * N, 256, 0, stream>>>(x, xn, D);
  k_transpose_cast<<<dim3(D / 32, N / 32, B), 256, 0, stream>>>(x, xT, N, D);

  for (int b = 0; b < B; ++b) {
    const bf16_t* xnb = xn + (size_t)b * N * D;
    const bf16_t* xTb = xT + (size_t)b * N * D;
    k_gemm_sim<<<dim3(N / 128, N / 128), 256, 0, stream>>>(xnb, sim, N, D);
    k_row_stats<<<N, 256, 0, stream>>>(sim, rmax, rsmi, N);
    k_dualize<<<(N / 256) * N, 256, 0, stream>>>(sim, rmax, rsmi, dual, N);
    k_gemm_neg<<<dim3(D / 128, N / 128), 256, 0, stream>>>(dual, xTb, xneg, N, D);
    k_residual_ln<<<N, 256, 0, stream>>>(x + (size_t)b * N * D, xneg, gamma, beta,
                                         out + (size_t)b * N * D, D);
  }
}
